// Attention_69277822485182
// MI455X (gfx1250) — matrix operand order repaired and checked by reading, not compiled, not run on hardware
//
#include <hip/hip_runtime.h>

// -------- static problem configuration (matches reference) --------
#define NSEQ     8
#define NH       32
#define HKV      8
#define GRP      4      // NH / HKV
#define HD       128    // head dim
#define QTOK     64     // new tokens per seq
#define MEMBLK   64     // tokens per cache block
#define MAXBLKS  32     // context blocks per seq
#define NTILES   33     // 32 context tiles + 1 new-token tile (64 keys each)

typedef __attribute__((ext_vector_type(16))) __bf16 v16bf;
typedef __attribute__((ext_vector_type(8)))  float  v8f;

// pack two f32 -> packed bf16 dword (lowers to v_cvt_pk_bf16_f32)
static __device__ inline unsigned pk2(float a, float b) {
    union { __bf16 h[2]; unsigned u; } x;
    x.h[0] = (__bf16)a; x.h[1] = (__bf16)b;
    return x.u;
}

// log2(e) / sqrt(128): softmax scale folded into exp2
#define CEXP (1.4426950408889634f * 0.08838834764831845f)

__global__ __launch_bounds__(128) void attn_fa_kernel(
    const float* __restrict__ qin,
    const float* __restrict__ kin,
    const float* __restrict__ vin,
    const float* __restrict__ k_cache,
    const float* __restrict__ v_cache,
    const int*   __restrict__ block_tables,
    float*       __restrict__ out)
{
    // K tile: 64 keys x 128 dims bf16, row stride 68 dwords (bank-conflict pad)
    __shared__ unsigned Kb[64 * 68];
    // V^T tile: 128 dims x 64 keys bf16, row stride 36 dwords
    __shared__ unsigned VTb[128 * 36];
    // per-wave P scratch: 16 rows x 64 keys bf16, row stride 36 dwords
    __shared__ unsigned Pb[4 * 16 * 36];

    const int b   = blockIdx.x;          // (s*HKV + kvh)*GRP + g
    const int s   = b >> 5;
    const int kvh = (b >> 2) & 7;
    const int g   = b & 3;
    const int qh  = kvh * GRP + g;

    const int tid  = threadIdx.x;
    const int w    = tid >> 5;           // wave id: owns query rows 16w..16w+15
    const int lane = tid & 31;
    const int ln   = lane & 15;
    const int hi   = lane >> 4;

    // staging thread mappings (constant across tiles)
    const int kk  = tid >> 1, hf = tid & 1;     // K stage: key, dim-half
    const int kq2 = tid & 31, dseg = tid >> 5;  // V stage: key pair, dim segment

    // ---- Q tile -> bf16 WMMA A layout, kept in VGPRs (4 chunks of K=32) ----
    v16bf a[4];
    {
        const float* qrow = qin + ((size_t)(s * QTOK + w * 16 + ln) * NH + qh) * HD;
        #pragma unroll
        for (int c = 0; c < 4; ++c) {
            const int d1 = c * 32 + hi * 8;
            const int d2 = c * 32 + 16 + hi * 8;
            float4 f0 = *(const float4*)(qrow + d1);
            float4 f1 = *(const float4*)(qrow + d1 + 4);
            float4 f2 = *(const float4*)(qrow + d2);
            float4 f3 = *(const float4*)(qrow + d2 + 4);
            union { unsigned u[8]; v16bf v; } t;
            t.u[0] = pk2(f0.x, f0.y); t.u[1] = pk2(f0.z, f0.w);
            t.u[2] = pk2(f1.x, f1.y); t.u[3] = pk2(f1.z, f1.w);
            t.u[4] = pk2(f2.x, f2.y); t.u[5] = pk2(f2.z, f2.w);
            t.u[6] = pk2(f3.x, f3.y); t.u[7] = pk2(f3.z, f3.w);
            a[c] = t.v;
        }
    }

    // ---- flash-attention state (C-layout rows 8*hi + r) ----
    v8f O[8];
    #pragma unroll
    for (int i = 0; i < 8; ++i) O[i] = (v8f)0.0f;
    float m[8], l[8];
    #pragma unroll
    for (int r = 0; r < 8; ++r) { m[r] = -__builtin_inff(); l[r] = 0.0f; }

    const int* bt = block_tables + s * MAXBLKS;

    for (int t = 0; t < NTILES; ++t) {
        __syncthreads();   // previous tile fully consumed before restaging

        const float* kp; const float* vp; size_t base;
        if (t < MAXBLKS) {
            base = (size_t)bt[t] * (size_t)(MEMBLK * HKV * HD) + (size_t)kvh * HD;
            kp = k_cache; vp = v_cache;
        } else {
            base = (size_t)(s * QTOK) * (HKV * HD) + (size_t)kvh * HD;
            kp = kin; vp = vin;
        }

        // stage K tile (row-major bf16)
        {
            const float* src = kp + base + (size_t)kk * (HKV * HD) + hf * 64;
            unsigned* dst = &Kb[kk * 68 + hf * 32];
            #pragma unroll
            for (int i = 0; i < 16; ++i) {
                float4 f = *(const float4*)(src + 4 * i);
                dst[2 * i]     = pk2(f.x, f.y);
                dst[2 * i + 1] = pk2(f.z, f.w);
            }
        }
        // stage V^T (bf16, keys contiguous)
        {
            const float* s0 = vp + base + (size_t)(2 * kq2) * (HKV * HD) + dseg * 32;
            const float* s1 = s0 + (HKV * HD);
            #pragma unroll
            for (int i = 0; i < 8; ++i) {
                float4 f0 = *(const float4*)(s0 + 4 * i);
                float4 f1 = *(const float4*)(s1 + 4 * i);
                const int dim = dseg * 32 + 4 * i;
                VTb[(dim + 0) * 36 + kq2] = pk2(f0.x, f1.x);
                VTb[(dim + 1) * 36 + kq2] = pk2(f0.y, f1.y);
                VTb[(dim + 2) * 36 + kq2] = pk2(f0.z, f1.z);
                VTb[(dim + 3) * 36 + kq2] = pk2(f0.w, f1.w);
            }
        }

        // prefetch next tile's staging sources into cache (global_prefetch_b8)
        if (t + 1 < NTILES) {
            const float* kpn; const float* vpn; size_t basen;
            if (t + 1 < MAXBLKS) {
                basen = (size_t)bt[t + 1] * (size_t)(MEMBLK * HKV * HD) + (size_t)kvh * HD;
                kpn = k_cache; vpn = v_cache;
            } else {
                basen = (size_t)(s * QTOK) * (HKV * HD) + (size_t)kvh * HD;
                kpn = kin; vpn = vin;
            }
            const float* ksrc = kpn + basen + (size_t)kk * (HKV * HD) + hf * 64;
            __builtin_prefetch(ksrc, 0, 1);
            __builtin_prefetch(ksrc + 32, 0, 1);
            const float* v0 = vpn + basen + (size_t)(2 * kq2) * (HKV * HD) + dseg * 32;
            __builtin_prefetch(v0, 0, 1);
            __builtin_prefetch(v0 + (HKV * HD), 0, 1);
        }
        __syncthreads();

        // ---- S = Q K^T : batch 4 B operands per D-chunk, then 4 WMMAs ----
        v8f st[4];
        #pragma unroll
        for (int j = 0; j < 4; ++j) st[j] = (v8f)0.0f;
        #pragma unroll
        for (int c = 0; c < 4; ++c) {
            union { uint4 u[2]; v16bf v; } bb[4];
            #pragma unroll
            for (int j = 0; j < 4; ++j) {
                const uint4* p4 = (const uint4*)&Kb[(16 * j + ln) * 68 + 16 * c + 4 * hi];
                bb[j].u[0] = p4[0]; bb[j].u[1] = p4[2];
            }
            #pragma unroll
            for (int j = 0; j < 4; ++j)
                st[j] = __builtin_amdgcn_wmma_f32_16x16x32_bf16(
                            false, a[c], false, bb[j].v, (short)0, st[j], false, false);
        }

        // ---- online softmax: row max over this 64-key block ----
        float tmax[8];
        #pragma unroll
        for (int r = 0; r < 8; ++r)
            tmax[r] = fmaxf(fmaxf(st[0][r], st[1][r]), fmaxf(st[2][r], st[3][r]));
        #pragma unroll
        for (int off = 8; off >= 1; off >>= 1) {
            #pragma unroll
            for (int r = 0; r < 8; ++r)
                tmax[r] = fmaxf(tmax[r], __shfl_xor(tmax[r], off, 32));
        }
        float alpha[8];
        #pragma unroll
        for (int r = 0; r < 8; ++r) {
            const float mn = fmaxf(m[r], tmax[r]);
            alpha[r] = __builtin_amdgcn_exp2f((m[r] - mn) * CEXP);
            m[r] = mn;
        }
        #pragma unroll
        for (int i = 0; i < 8; ++i) {
            #pragma unroll
            for (int r = 0; r < 8; ++r) O[i][r] *= alpha[r];
        }

        // ---- P = exp2((S - m)*CEXP); row sums; write bf16 P to per-wave LDS ----
        float lsum[8];
        #pragma unroll
        for (int r = 0; r < 8; ++r) lsum[r] = 0.0f;
        unsigned short* Pus = (unsigned short*)&Pb[w * 576];
        #pragma unroll
        for (int j = 0; j < 4; ++j) {
            const int key = 16 * j + ln;
            #pragma unroll
            for (int r = 0; r < 8; ++r) {
                const float pv = __builtin_amdgcn_exp2f((st[j][r] - m[r]) * CEXP);
                lsum[r] += pv;
                union { __bf16 h; unsigned short us; } cv; cv.h = (__bf16)pv;
                Pus[(hi * 8 + r) * 72 + key] = cv.us;
            }
        }
        #pragma unroll
        for (int off = 8; off >= 1; off >>= 1) {
            #pragma unroll
            for (int r = 0; r < 8; ++r) lsum[r] += __shfl_xor(lsum[r], off, 32);
        }
        #pragma unroll
        for (int r = 0; r < 8; ++r) l[r] = l[r] * alpha[r] + lsum[r];

        // ---- O += P V : batch 4 V^T operands per group, then 4 WMMAs ----
        #pragma unroll
        for (int h2 = 0; h2 < 2; ++h2) {
            union { uint4 u[2]; v16bf v; } pa;   // P in A layout (same-wave in-order DS)
            pa.u[0] = *(const uint4*)&Pus[ln * 72 + 32 * h2 + 8 * hi];
            pa.u[1] = *(const uint4*)&Pus[ln * 72 + 32 * h2 + 16 + 8 * hi];
            #pragma unroll
            for (int tg = 0; tg < 2; ++tg) {
                union { uint4 u[2]; v16bf v; } vb[4];
                #pragma unroll
                for (int u = 0; u < 4; ++u) {
                    const int dim = 16 * (4 * tg + u) + ln;
                    const uint4* vv = (const uint4*)&VTb[dim * 36 + 16 * h2 + 4 * hi];
                    vb[u].u[0] = vv[0]; vb[u].u[1] = vv[2];
                }
                #pragma unroll
                for (int u = 0; u < 4; ++u) {
                    const int tt = 4 * tg + u;
                    O[tt] = __builtin_amdgcn_wmma_f32_16x16x32_bf16(
                                false, pa.v, false, vb[u].v, (short)0, O[tt], false, false);
                }
            }
        }
    }

    // ---- epilogue: normalize and store (C layout -> [token, head, dim]) ----
    float inv[8];
    #pragma unroll
    for (int r = 0; r < 8; ++r) inv[r] = 1.0f / l[r];
    #pragma unroll
    for (int tt = 0; tt < 8; ++tt) {
        #pragma unroll
        for (int r = 0; r < 8; ++r) {
            const int row = hi * 8 + r;
            const int tok = s * QTOK + w * 16 + row;
            out[((size_t)tok * NH + qh) * HD + 16 * tt + ln] = O[tt][r] * inv[r];
        }
    }
}

extern "C" void kernel_launch(void* const* d_in, const int* in_sizes, int n_in,
                              void* d_out, int out_size, void* d_ws, size_t ws_size,
                              hipStream_t stream) {
    (void)in_sizes; (void)n_in; (void)out_size; (void)d_ws; (void)ws_size;
    const float* q        = (const float*)d_in[0];
    const float* k        = (const float*)d_in[1];
    const float* v        = (const float*)d_in[2];
    const float* k_cache  = (const float*)d_in[3];
    const float* v_cache  = (const float*)d_in[4];
    const int*   bt       = (const int*)d_in[5];
    // d_in[6] (slot_mapping) unused: scattered slots are disjoint from the
    // blocks read via block_tables, and the caches are inputs we must not mutate.
    float* out = (float*)d_out;

    dim3 grid(NSEQ * HKV * GRP);   // 256 workgroups, GQA-sharing WGs adjacent for L2 reuse
    dim3 block(128);               // 4 wave32; each wave owns 16 query rows
    attn_fa_kernel<<<grid, block, 0, stream>>>(q, k, v, k_cache, v_cache, bt, out);
}
